// CrossAttention2_86586540687656
// MI455X (gfx1250) — compile-verified, matrix operand order checked by reading, not run on hardware
//
#include <hip/hip_runtime.h>
#include <cstdint>

// ---------------------------------------------------------------------------
// CDNA5 (gfx1250) cross-attention, bf16 WMMA (16x16x32), f32 accumulate.
// Pipeline (all on `stream`):
//   0) pack weights f32->bf16; transpose-pack x/context -> [b][pos][chan] bf16
//   1) Q/K/V projections via WMMA GEMM (M=512,N=1024,K=512 per batch)
//        Q,K stored [b][h][pos][64]  (A/B-fragment friendly)
//        V   stored [b][h][64][pos]
//   2) softmax row stats (m, 1/l) per query row via WMMA score tiles
//   3) fused scores->exp->V@P via WMMA (attn matrix never hits memory);
//      score D-tiles repack in-register into B-fragments (layout identity)
//   4) output projection via the same WMMA GEMM, f32 epilogue to d_out
// ---------------------------------------------------------------------------

typedef __bf16 bf16_t;
typedef __attribute__((ext_vector_type(16))) __bf16 v16bf;
typedef __attribute__((ext_vector_type(8)))  __bf16 v8bf;
typedef __attribute__((ext_vector_type(8)))  float  v8f;

#define BSZ 8
#define CH 512
#define HWD 1024
#define NH 8
#define DH 64
#define INV_SCALE 0.125f   // 1/sqrt(64)

__device__ __forceinline__ v8f wmma_bf16(v16bf a, v16bf b, v8f c) {
  return __builtin_amdgcn_wmma_f32_16x16x32_bf16(false, a, false, b, (short)0, c,
                                                 false, false);
}

// A-frag (16xK=32, row-major [row][k], k contiguous) and B-frag
// (K=32x16 taken from memory stored [col][k], k contiguous) share the same
// per-lane gather: lane L holds row/col (L&15); element e maps to
// k = (L<16?0:8) + (e<8 ? e : 16 + e-8)  => two contiguous 16B loads.
__device__ __forceinline__ v16bf load_frag(const bf16_t* __restrict__ p, int ld,
                                           int rbase, int kbase, int lane) {
  const int row = rbase + (lane & 15);
  const int hi  = (lane >> 4) & 1;
  const bf16_t* base = p + (size_t)row * ld + kbase + hi * 8;
  union { v16bf f; v8bf h[2]; } u;
  u.h[0] = *(const v8bf*)(base);
  u.h[1] = *(const v8bf*)(base + 16);
  return u.f;
}

__device__ __forceinline__ v8bf cvt_v8(v8f a) {
  v8bf r;
#pragma unroll
  for (int i = 0; i < 8; ++i) r[i] = (bf16_t)a[i];
  return r;
}

// ------------------------------ pack kernels -------------------------------

__global__ __launch_bounds__(256) void pack_w_kernel(const float* __restrict__ w,
                                                     bf16_t* __restrict__ o, int n) {
  int i = blockIdx.x * 256 + threadIdx.x;
  if (i < n) o[i] = (bf16_t)w[i];
}

// x[b][c][s] f32 -> xt[b][s][c] bf16   (total = BSZ*CH*HWD = 2^22, exact grid)
__global__ __launch_bounds__(256) void pack_xT_kernel(const float* __restrict__ x,
                                                      bf16_t* __restrict__ xt) {
  size_t idx = (size_t)blockIdx.x * 256 + threadIdx.x;
  int s = (int)(idx & (HWD - 1));
  int c = (int)((idx >> 10) & (CH - 1));
  int b = (int)(idx >> 19);
  xt[((size_t)b * HWD + s) * CH + c] = (bf16_t)x[idx];
}

// ------------------------------- GEMM --------------------------------------
// Y[m,s] = sum_k A[m,k]*B[b][s,k] + bias[m]; M=512, N=1024, K=512.
// MODE 0: store bf16 [b][head][s][d]   (Q,K)
// MODE 1: store bf16 [b][head][d][s]   (V)
// MODE 2: store f32  [b][m][s]         (final output)
template <int MODE>
__global__ __launch_bounds__(256) void gemm512_kernel(const bf16_t* __restrict__ A,
                                                      const bf16_t* __restrict__ B,
                                                      const float* __restrict__ bias,
                                                      void* __restrict__ Out) {
  const int lane = threadIdx.x & 31;
  const int wave = threadIdx.x >> 5;
  const int m0 = blockIdx.y * 128 + wave * 16;
  const int n0 = blockIdx.x * 16;
  const int b  = blockIdx.z;
  const bf16_t* Bb = B + (size_t)b * HWD * CH;

  v8f acc = {};
#pragma unroll 4
  for (int k = 0; k < CH; k += 32) {
    v16bf af = load_frag(A, CH, m0, k, lane);
    v16bf bf = load_frag(Bb, CH, n0, k, lane);
    acc = wmma_bf16(af, bf, acc);
  }

  const int col  = n0 + (lane & 15);
  const int roff = ((lane >> 4) & 1) * 8;  // D-tile: lanes>=16 hold M = 8+r

  if (MODE == 0) {
    const int head = m0 >> 6;
    const int d0   = (m0 & 63) + roff;
    bf16_t* O = (bf16_t*)Out;
    v8bf pk;
#pragma unroll
    for (int r = 0; r < 8; ++r) pk[r] = (bf16_t)(acc[r] + bias[m0 + roff + r]);
    *(v8bf*)(O + ((((size_t)b * NH + head) * HWD) + col) * DH + d0) = pk;
  } else if (MODE == 1) {
    bf16_t* O = (bf16_t*)Out;
#pragma unroll
    for (int r = 0; r < 8; ++r) {
      int m = m0 + roff + r;
      O[(((size_t)b * NH + (m >> 6)) * DH + (m & 63)) * HWD + col] =
          (bf16_t)(acc[r] + bias[m]);
    }
  } else {
    float* O = (float*)Out;
#pragma unroll
    for (int r = 0; r < 8; ++r) {
      int m = m0 + roff + r;
      O[((size_t)b * CH + m) * HWD + col] = acc[r] + bias[m];
    }
  }
}

// --------------------------- softmax row stats -----------------------------
// Per (b,h): s[j,i] = (Q[j,:]·K[i,:]) * INV_SCALE; write ml[j] = (max_i, 1/sum_i exp).
__global__ __launch_bounds__(256) void stats_kernel(const bf16_t* __restrict__ Qt,
                                                    const bf16_t* __restrict__ Kt,
                                                    float2* __restrict__ ml) {
  const int lane = threadIdx.x & 31;
  const int wave = threadIdx.x >> 5;
  const int bh = blockIdx.z * NH + blockIdx.y;
  const bf16_t* Q = Qt + (size_t)bh * HWD * DH;
  const bf16_t* K = Kt + (size_t)bh * HWD * DH;
  const int j0 = blockIdx.x * 128 + wave * 16;

  v16bf a0 = load_frag(Q, DH, j0, 0, lane);
  v16bf a1 = load_frag(Q, DH, j0, 32, lane);

  float m8[8], l8[8];
#pragma unroll
  for (int r = 0; r < 8; ++r) { m8[r] = -1.0e30f; l8[r] = 0.0f; }

  for (int it = 0; it < HWD; it += 16) {
    v16bf b0 = load_frag(K, DH, it, 0, lane);
    v16bf b1 = load_frag(K, DH, it, 32, lane);
    v8f s = {};
    s = wmma_bf16(a0, b0, s);
    s = wmma_bf16(a1, b1, s);
#pragma unroll
    for (int r = 0; r < 8; ++r) {
      float v = s[r] * INV_SCALE;
      float mn = fmaxf(m8[r], v);
      l8[r] = l8[r] * __expf(m8[r] - mn) + __expf(v - mn);
      m8[r] = mn;
    }
  }
  // merge (m,l) across the 16 lanes holding one row
#pragma unroll
  for (int r = 0; r < 8; ++r) {
    float m = m8[r], l = l8[r];
    for (int off = 8; off >= 1; off >>= 1) {
      float mo = __shfl_xor(m, off, 16);
      float lo = __shfl_xor(l, off, 16);
      float mn = fmaxf(m, mo);
      l = l * __expf(m - mn) + lo * __expf(mo - mn);
      m = mn;
    }
    m8[r] = m; l8[r] = l;
  }
  if ((lane & 15) == 0) {
    int jb = j0 + ((lane >> 4) & 1) * 8;
#pragma unroll
    for (int r = 0; r < 8; ++r)
      ml[(size_t)bh * HWD + jb + r] = make_float2(m8[r], 1.0f / l8[r]);
  }
}

// ----------------------- fused scores->softmax->V@P ------------------------
// out[d,i] = sum_j V[d,j] * exp(s[j,i]*INV_SCALE - m[j]) * rl[j]
// Score D-tiles (M=j,N=i) repack in-register into a 32-deep B-fragment:
//   B elem e<8  <- tile(j=jt..)   value r=e     (same lane, same column)
//   B elem e>=8 <- tile(j=jt+16..) value r=e-8
__global__ __launch_bounds__(256) void attn_kernel(const bf16_t* __restrict__ Qt,
                                                   const bf16_t* __restrict__ Kt,
                                                   const bf16_t* __restrict__ Vv,
                                                   const float2* __restrict__ ml,
                                                   bf16_t* __restrict__ Ot) {
  const int lane = threadIdx.x & 31;
  const int wave = threadIdx.x >> 5;
  const int n = blockIdx.y;
  const int b = blockIdx.z;
  const int bh = b * NH + n;
  const bf16_t* Q = Qt + (size_t)bh * HWD * DH;
  const bf16_t* K = Kt + (size_t)bh * HWD * DH;
  const bf16_t* V = Vv + (size_t)bh * DH * HWD;
  const float2* mlb = ml + (size_t)bh * HWD;
  const int i0 = blockIdx.x * 128 + wave * 16;
  const int roff = ((lane >> 4) & 1) * 8;

  // K columns i0..i0+15 stay resident in registers
  v16bf kb0 = load_frag(K, DH, i0, 0, lane);
  v16bf kb1 = load_frag(K, DH, i0, 32, lane);

  v8f acc[4] = {{}, {}, {}, {}};

  for (int jt = 0; jt < HWD; jt += 32) {
    v16bf qa0 = load_frag(Q, DH, jt, 0, lane);
    v16bf qa1 = load_frag(Q, DH, jt, 32, lane);
    v16bf qb0 = load_frag(Q, DH, jt + 16, 0, lane);
    v16bf qb1 = load_frag(Q, DH, jt + 16, 32, lane);

    v8f s0 = {};
    s0 = wmma_bf16(qa0, kb0, s0);
    s0 = wmma_bf16(qa1, kb1, s0);
    v8f s1 = {};
    s1 = wmma_bf16(qb0, kb0, s1);
    s1 = wmma_bf16(qb1, kb1, s1);

    v16bf pf;
#pragma unroll
    for (int r = 0; r < 8; ++r) {
      float2 m0 = mlb[jt + roff + r];
      float2 m1 = mlb[jt + 16 + roff + r];
      float p0 = __expf(s0[r] * INV_SCALE - m0.x) * m0.y;
      float p1 = __expf(s1[r] * INV_SCALE - m1.x) * m1.y;
      pf[r]     = (bf16_t)p0;
      pf[r + 8] = (bf16_t)p1;
    }

#pragma unroll
    for (int t = 0; t < 4; ++t) {
      v16bf va = load_frag(V, HWD, t * 16, jt, lane);  // V rows d, k = j
      acc[t] = wmma_bf16(va, pf, acc[t]);
    }
  }

  const int i = i0 + (lane & 15);
  bf16_t* Orow = Ot + (((size_t)b * HWD + i) * CH) + n * DH;
#pragma unroll
  for (int t = 0; t < 4; ++t)
    *(v8bf*)(Orow + t * 16 + roff) = cvt_v8(acc[t]);
}

// ------------------------------- launcher ----------------------------------

extern "C" void kernel_launch(void* const* d_in, const int* in_sizes, int n_in,
                              void* d_out, int out_size, void* d_ws, size_t ws_size,
                              hipStream_t stream) {
  (void)in_sizes; (void)n_in; (void)out_size; (void)ws_size;
  const float* x   = (const float*)d_in[0];
  const float* ctx = (const float*)d_in[1];
  const float* Wq  = (const float*)d_in[2];
  const float* bq  = (const float*)d_in[3];
  const float* Wk  = (const float*)d_in[4];
  const float* bk  = (const float*)d_in[5];
  const float* Wv  = (const float*)d_in[6];
  const float* bv  = (const float*)d_in[7];
  const float* Wo  = (const float*)d_in[8];
  const float* bo  = (const float*)d_in[9];

  const size_t MB = 1ull << 20;
  uint8_t* ws = (uint8_t*)d_ws;
  bf16_t* Qt  = (bf16_t*)(ws + 0 * MB);    // [b][h][pos][64]  8 MB
  bf16_t* Kt  = (bf16_t*)(ws + 8 * MB);    // [b][h][pos][64]  8 MB
  bf16_t* Vv  = (bf16_t*)(ws + 16 * MB);   // [b][h][64][pos]  8 MB
  bf16_t* Xt  = (bf16_t*)(ws + 24 * MB);   // [b][pos][ch]     8 MB (reused as Ot)
  bf16_t* Ct  = (bf16_t*)(ws + 32 * MB);   // [b][pos][ch]     8 MB
  bf16_t* Wqb = (bf16_t*)(ws + 40 * MB);
  bf16_t* Wkb = (bf16_t*)(ws + 40 * MB + 512 * 1024);
  bf16_t* Wvb = (bf16_t*)(ws + 41 * MB);
  bf16_t* Wob = (bf16_t*)(ws + 41 * MB + 512 * 1024);
  float2* ml  = (float2*)(ws + 42 * MB);   // [b*h][pos]       512 KB
  bf16_t* Ot  = Xt;                        // Xt dead after Q projection

  const int NW = CH * CH;                  // 262144
  pack_w_kernel<<<(NW + 255) / 256, 256, 0, stream>>>(Wq, Wqb, NW);
  pack_w_kernel<<<(NW + 255) / 256, 256, 0, stream>>>(Wk, Wkb, NW);
  pack_w_kernel<<<(NW + 255) / 256, 256, 0, stream>>>(Wv, Wvb, NW);
  pack_w_kernel<<<(NW + 255) / 256, 256, 0, stream>>>(Wo, Wob, NW);

  const int NT = BSZ * CH * HWD / 256;     // 16384 blocks, exact cover
  pack_xT_kernel<<<NT, 256, 0, stream>>>(x, Xt);
  pack_xT_kernel<<<NT, 256, 0, stream>>>(ctx, Ct);

  dim3 gg(HWD / 16, CH / 128, BSZ);        // (64, 4, 8)
  gemm512_kernel<0><<<gg, 256, 0, stream>>>(Wqb, Xt, bq, (void*)Qt);
  gemm512_kernel<0><<<gg, 256, 0, stream>>>(Wkb, Ct, bk, (void*)Kt);
  gemm512_kernel<1><<<gg, 256, 0, stream>>>(Wvb, Ct, bv, (void*)Vv);

  dim3 ga(HWD / 128, NH, BSZ);             // (8, 8, 8)
  stats_kernel<<<ga, 256, 0, stream>>>(Qt, Kt, ml);
  attn_kernel<<<ga, 256, 0, stream>>>(Qt, Kt, Vv, ml, Ot);

  gemm512_kernel<2><<<gg, 256, 0, stream>>>(Wob, Ot, bo, d_out);
}
